// RelPosAttention_38268158607448
// MI455X (gfx1250) — compile-verified, matrix operand order checked by reading, not run on hardware
//
#include <hip/hip_runtime.h>
#include <hip/hip_bf16.h>
#include <stddef.h>

// ---------------------------------------------------------------------------
// RelPosAttention for MI455X (gfx1250): bf16 WMMA + async global->LDS pipeline.
// B=1, H=W=64, N=4096, C=768, heads=12, hd=64.
// ---------------------------------------------------------------------------

typedef __attribute__((ext_vector_type(16))) __bf16         v16bf;
typedef __attribute__((ext_vector_type(8)))  unsigned int   uint8v;
typedef __attribute__((ext_vector_type(8)))  float          v8f;

#define NHEADS 12
#define N_TOK  4096
#define HD     64
#define CDIM   768

static __device__ __forceinline__ unsigned short f2bf(float f) {
    unsigned int u = __float_as_uint(f);
    unsigned int r = (u + 0x7FFFu + ((u >> 16) & 1u)) >> 16;  // RNE
    return (unsigned short)r;
}

static __device__ __forceinline__ v8f vzero8() {
    v8f z;
#pragma unroll
    for (int i = 0; i < 8; ++i) z[i] = 0.0f;
    return z;
}

// Async global->LDS copy (CDNA5): 16B per lane, tracked by ASYNCcnt.
// LDS offset = low 32 bits of the generic shared-memory address.
static __device__ __forceinline__ void async_copy_b128(void* lds_ptr, const void* gptr) {
    unsigned lds = (unsigned)(size_t)lds_ptr;
    unsigned long long g = (unsigned long long)gptr;
    asm volatile("global_load_async_to_lds_b128 %0, %1, off"
                 :: "v"(lds), "v"(g) : "memory");
}
static __device__ __forceinline__ void wait_async0() {
    asm volatile("s_wait_asynccnt 0" ::: "memory");
}

// K index packing of the 16x32 bf16 A layout (and symmetric B layout):
// vgpr pair v (0..7) holds k = {base, base+1}, base = (v<4 ? 2v : 2v+8) + half*8
static __device__ __forceinline__ int kpair_base(int v, int half) {
    return (v < 4 ? 2 * v : 2 * v + 8) + half * 8;
}

// A fragment 16(M)x32(K): p -> element [m=0][k=0], row stride ld (bf16 elems).
// Per lane the 16 elements form two contiguous 16B chunks -> 2x ds_load_b128.
static __device__ __forceinline__ v16bf frag_a(const unsigned short* p, int ld, int lane) {
    const int m = lane & 15, half = lane >> 4;
    uint8v r;
#pragma unroll
    for (int v = 0; v < 8; ++v) {
        int k = kpair_base(v, half);
        r[v] = *(const unsigned int*)(p + m * ld + k);
    }
    return __builtin_bit_cast(v16bf, r);
}

// B fragment 32(K)x16(N) where the source is stored N-major: B[k][n] = p[n*ld+k].
static __device__ __forceinline__ v16bf frag_bT(const unsigned short* p, int ld, int lane) {
    const int n = lane & 15, half = lane >> 4;
    uint8v r;
#pragma unroll
    for (int v = 0; v < 8; ++v) {
        int k = kpair_base(v, half);
        r[v] = *(const unsigned int*)(p + n * ld + k);
    }
    return __builtin_bit_cast(v16bf, r);
}

static __device__ __forceinline__ v8f wmma_bf16(v16bf a, v16bf b, v8f c) {
    return __builtin_amdgcn_wmma_f32_16x16x32_bf16(false, a, false, b, (short)0, c,
                                                   false, false);
}

// C/D layout: VGPR r holds row (r + half*8), column = lane&15.

// ---------------------------------------------------------------------------
// Kernel 1: fp32 -> bf16 (raw ushort) conversion for weight tables
// ---------------------------------------------------------------------------
__global__ void cvt_bf16_kernel(const float* __restrict__ src,
                                unsigned short* __restrict__ dst, int n) {
    int i = blockIdx.x * 256 + threadIdx.x;
    if (i < n) dst[i] = f2bf(src[i]);
}

// ---------------------------------------------------------------------------
// Kernel 2: qkv GEMM. X(4096x768 f32) @ Wbf(768x2304 bf16) + bias.
// Epilogue scatters to Qbf/Kbf laid out [head][n][hd] and V TRANSPOSED into
// Vt[head][hd][n] so the attention V tile is an async-copyable byte range.
// ---------------------------------------------------------------------------
__global__ __launch_bounds__(256) void gemm_qkv_kernel(
    const float* __restrict__ X, const unsigned short* __restrict__ Wbf,
    const float* __restrict__ bias, unsigned short* __restrict__ Qbf,
    unsigned short* __restrict__ Kbf, unsigned short* __restrict__ Vt) {
    __shared__ __align__(16) unsigned short As[64][40];   // [m][k]
    __shared__ __align__(16) unsigned short BsT[64][40];  // [n][k] transposed
    const int tid = threadIdx.x, lane = tid & 31, wave = tid >> 5;
    const int bm = blockIdx.y * 64, bn = blockIdx.x * 64;
    const int mOff = (wave & 3) * 16, nOff = (wave >> 2) * 32;
    v8f c0 = vzero8(), c1 = vzero8();

    for (int k0 = 0; k0 < CDIM; k0 += 32) {
        __syncthreads();
#pragma unroll
        for (int i = 0; i < 8; ++i) {  // A tile: 64x32 fp32 -> bf16
            int lin = tid + i * 256;
            int r = lin >> 5, c = lin & 31;
            As[r][c] = f2bf(X[(bm + r) * CDIM + k0 + c]);
        }
#pragma unroll
        for (int i = 0; i < 8; ++i) {  // B tile 32x64, stored transposed [n][k]
            int lin = tid + i * 256;
            int r = lin >> 6, c = lin & 63;
            BsT[c][r] = Wbf[(k0 + r) * (3 * CDIM) + bn + c];
        }
        __syncthreads();
        v16bf a  = frag_a(&As[mOff][0], 40, lane);
        v16bf b0 = frag_bT(&BsT[nOff][0], 40, lane);
        v16bf b1 = frag_bT(&BsT[nOff + 16][0], 40, lane);
        c0 = wmma_bf16(a, b0, c0);
        c1 = wmma_bf16(a, b1, c1);
    }
    const int m15 = lane & 15, half = lane >> 4;
    const int colHi = bn >> 6;          // block-uniform
    const int three = colHi / 12;
    const int head = colHi - three * 12;
#pragma unroll
    for (int f = 0; f < 2; ++f) {
#pragma unroll
        for (int r = 0; r < 8; ++r) {
            int row = bm + mOff + r + half * 8;
            int d = nOff + f * 16 + m15;  // col & 63
            float v = ((f == 0) ? c0[r] : c1[r]) + bias[bn + d];
            unsigned short bv = f2bf(v);
            if (three == 0)      Qbf[((head << 12) + row) * HD + d] = bv;
            else if (three == 1) Kbf[((head << 12) + row) * HD + d] = bv;
            else                 Vt[((size_t)(head << 6) + d) * N_TOK + row] = bv;
        }
    }
}

// ---------------------------------------------------------------------------
// Kernel 3: decomposed rel-pos bias GEMMs.
// mode 0: rel_h, stored TRANSPOSED  relT[head][k][n] ;  n = i*64 + w
// mode 1: rel_w, stored             rel [head][n][k] ;  n = h*64 + i
// ---------------------------------------------------------------------------
__global__ __launch_bounds__(256) void relpos_kernel(
    const unsigned short* __restrict__ Qbf, const unsigned short* __restrict__ tbl,
    float* __restrict__ rel, int mode) {
    __shared__ __align__(16) unsigned short Qs[64][72];   // [row][c]
    __shared__ __align__(16) unsigned short BtT[64][72];  // [kcol][c] (N-major)
    const int tid = threadIdx.x, lane = tid & 31, wave = tid >> 5;
    const int i = blockIdx.x, head = blockIdx.y;
    const unsigned short* Qh = Qbf + (size_t)head * N_TOK * HD;

#pragma unroll
    for (int t = 0; t < 16; ++t) {
        int lin = tid + t * 256;
        int r = lin >> 6, c = lin & 63;
        int grow = (mode == 0) ? (i * 64 + r) : (r * 64 + i);
        Qs[r][c] = Qh[grow * HD + c];
    }
#pragma unroll
    for (int t = 0; t < 16; ++t) {
        int lin = tid + t * 256;
        int kc = lin >> 6, c = lin & 63;
        BtT[kc][c] = tbl[(i - kc + 63) * HD + c];
    }
    __syncthreads();

    const int mOff = (wave & 3) * 16, nOff = (wave >> 2) * 32;
    const int m15 = lane & 15, half = lane >> 4;
    v16bf a0 = frag_a(&Qs[mOff][0], 72, lane);
    v16bf a1 = frag_a(&Qs[mOff][32], 72, lane);
#pragma unroll
    for (int f = 0; f < 2; ++f) {
        v8f c = vzero8();
        v16bf b0 = frag_bT(&BtT[nOff + f * 16][0], 72, lane);
        v16bf b1 = frag_bT(&BtT[nOff + f * 16][32], 72, lane);
        c = wmma_bf16(a0, b0, c);
        c = wmma_bf16(a1, b1, c);
#pragma unroll
        for (int r = 0; r < 8; ++r) {
            int rl = mOff + r + half * 8;
            int k = nOff + f * 16 + m15;
            if (mode == 0) {
                int n = i * 64 + rl;
                rel[((size_t)head * 64 + k) * N_TOK + n] = c[r];
            } else {
                int n = rl * 64 + i;
                rel[((size_t)head * N_TOK + n) * HD + k] = c[r];
            }
        }
    }
}

// ---------------------------------------------------------------------------
// Kernel 4: fused flash attention per head, double-buffered async-LDS pipeline.
// Block = (qStrip of 128 rows, head). 8 waves; wave owns 16 query rows.
// K tile, pre-transposed V tile and the rel_h column slice stream into LDS via
// global_load_async_to_lds_b128 while WMMAs run on the other buffer.
// ---------------------------------------------------------------------------
__global__ __launch_bounds__(256) void attn_kernel(
    const unsigned short* __restrict__ Qbf, const unsigned short* __restrict__ Kbf,
    const unsigned short* __restrict__ Vt, const float* __restrict__ relhT,
    const float* __restrict__ relw, float* __restrict__ Obuf) {
    __shared__ __align__(16) unsigned short Ks[2][64][72];   // [key][dim]
    __shared__ __align__(16) unsigned short Vs[2][64][72];   // [dim][key] (N-major)
    __shared__ __align__(16) unsigned short Ps[8][16][72];   // per-wave P staging
    __shared__ __align__(16) float rhS[2][128];              // rel_h column slice

    const int tid = threadIdx.x, lane = tid & 31, wave = tid >> 5;
    const int head = blockIdx.y, qBase = blockIdx.x * 128;
    const unsigned short* Qh = Qbf + (size_t)head * N_TOK * HD;
    const unsigned short* Kh = Kbf + (size_t)head * N_TOK * HD;
    const unsigned short* Vth = Vt + (size_t)head * HD * N_TOK;
    const float* rhT = relhT + (size_t)head * 64 * N_TOK;
    const float* rwh = relw + (size_t)head * N_TOK * HD;

    const int m15 = lane & 15, half = lane >> 4;
    const int qRow0 = qBase + wave * 16;
    const float scale = 0.125f;  // 64^-0.5

    // issue async copies for key tile t into buffer b (4x b128 chunks/thread)
    auto issue_tile = [&](int t, int b) {
#pragma unroll
        for (int j = 0; j < 2; ++j) {   // K tile: 64 rows x 128B = 512 chunks
            int ck = tid + j * 256;
            int r = ck >> 3, c8 = (ck & 7) << 3;
            async_copy_b128(&Ks[b][r][c8], Kh + ((t << 6) + r) * HD + c8);
        }
#pragma unroll
        for (int j = 0; j < 2; ++j) {   // V tile (rows = hd, cols = keys)
            int ck = tid + j * 256;
            int r = ck >> 3, c8 = (ck & 7) << 3;
            async_copy_b128(&Vs[b][r][c8], Vth + (size_t)r * N_TOK + (t << 6) + c8);
        }
        if (tid < 32)                   // rel_h slice: 128 f32 = 32 chunks
            async_copy_b128(&rhS[b][tid * 4], rhT + (size_t)t * N_TOK + qBase + tid * 4);
    };

    v16bf aq0 = frag_a(Qh + (size_t)qRow0 * HD, HD, lane);       // K dims 0..31
    v16bf aq1 = frag_a(Qh + (size_t)qRow0 * HD + 32, HD, lane);  // K dims 32..63

    // key-invariant rel_w bias in D-fragment layout (loaded once)
    float bw[4][8];
#pragma unroll
    for (int ci = 0; ci < 4; ++ci)
#pragma unroll
        for (int r = 0; r < 8; ++r)
            bw[ci][r] = rwh[(size_t)(qRow0 + r + half * 8) * HD + ci * 16 + m15];

    float mrow[8], lrow[8];
#pragma unroll
    for (int r = 0; r < 8; ++r) { mrow[r] = -3.0e38f; lrow[r] = 0.0f; }
    v8f o[4];
#pragma unroll
    for (int hi = 0; hi < 4; ++hi) o[hi] = vzero8();

    issue_tile(0, 0);

    for (int kt = 0; kt < 64; ++kt) {
        const int buf = kt & 1;
        wait_async0();       // own async copies for tile kt complete
        __syncthreads();     // everyone's copies complete; prev buffer free
        if (kt + 1 < 64) issue_tile(kt + 1, buf ^ 1);

        // S = Q * K^T : 16x64 strip per wave
        v8f s[4];
#pragma unroll
        for (int ci = 0; ci < 4; ++ci) {
            s[ci] = vzero8();
            v16bf b0 = frag_bT(&Ks[buf][ci * 16][0], 72, lane);
            v16bf b1 = frag_bT(&Ks[buf][ci * 16][32], 72, lane);
            s[ci] = wmma_bf16(aq0, b0, s[ci]);
            s[ci] = wmma_bf16(aq1, b1, s[ci]);
        }

        // scale + bias (rel_h from LDS slice, rel_w from hoisted registers)
        float rh[8];
#pragma unroll
        for (int r = 0; r < 8; ++r) rh[r] = rhS[buf][wave * 16 + r + half * 8];
#pragma unroll
        for (int ci = 0; ci < 4; ++ci)
#pragma unroll
            for (int r = 0; r < 8; ++r)
                s[ci][r] = s[ci][r] * scale + rh[r] + bw[ci][r];

        // online softmax (row lives across the 16 lanes of one half-wave)
        float corr[8];
#pragma unroll
        for (int r = 0; r < 8; ++r) {
            float t = fmaxf(fmaxf(s[0][r], s[1][r]), fmaxf(s[2][r], s[3][r]));
#pragma unroll
            for (int d = 1; d < 16; d <<= 1) t = fmaxf(t, __shfl_xor(t, d, 32));
            float mn = fmaxf(mrow[r], t);
            corr[r] = __expf(mrow[r] - mn);
            mrow[r] = mn;
            float rs = 0.0f;
#pragma unroll
            for (int ci = 0; ci < 4; ++ci) {
                float p = __expf(s[ci][r] - mn);
                s[ci][r] = p;
                rs += p;
            }
#pragma unroll
            for (int d = 1; d < 16; d <<= 1) rs += __shfl_xor(rs, d, 32);
            lrow[r] = lrow[r] * corr[r] + rs;
        }
#pragma unroll
        for (int hi = 0; hi < 4; ++hi)
#pragma unroll
            for (int r = 0; r < 8; ++r) o[hi][r] *= corr[r];

        // stage P (D-frag layout -> row-major LDS -> A-frag layout), wave-private
#pragma unroll
        for (int ci = 0; ci < 4; ++ci)
#pragma unroll
            for (int r = 0; r < 8; ++r)
                Ps[wave][r + half * 8][ci * 16 + m15] = f2bf(s[ci][r]);

        v16bf ap0 = frag_a(&Ps[wave][0][0], 72, lane);
        v16bf ap1 = frag_a(&Ps[wave][0][32], 72, lane);
#pragma unroll
        for (int hi = 0; hi < 4; ++hi) {  // O += P * V
            v16bf b0 = frag_bT(&Vs[buf][hi * 16][0], 72, lane);
            v16bf b1 = frag_bT(&Vs[buf][hi * 16][32], 72, lane);
            o[hi] = wmma_bf16(ap0, b0, o[hi]);
            o[hi] = wmma_bf16(ap1, b1, o[hi]);
        }
    }

    // normalize and write to (n, head*64+d) fp32 layout
#pragma unroll
    for (int hi = 0; hi < 4; ++hi)
#pragma unroll
        for (int r = 0; r < 8; ++r) {
            int row = qRow0 + r + half * 8;
            float l = lrow[r];
            l = (l > 0.0f) ? l : 1.0f;
            Obuf[(size_t)row * CDIM + head * HD + hi * 16 + m15] = o[hi][r] / l;
        }
}

// ---------------------------------------------------------------------------
// Kernel 5: output projection. Obuf(4096x768 f32) @ Wproj(768x768 bf16) + b.
// ---------------------------------------------------------------------------
__global__ __launch_bounds__(256) void gemm_proj_kernel(
    const float* __restrict__ A, const unsigned short* __restrict__ Wbf,
    const float* __restrict__ bias, float* __restrict__ out) {
    __shared__ __align__(16) unsigned short As[64][40];
    __shared__ __align__(16) unsigned short BsT[64][40];  // [n][k]
    const int tid = threadIdx.x, lane = tid & 31, wave = tid >> 5;
    const int bm = blockIdx.y * 64, bn = blockIdx.x * 64;
    const int mOff = (wave & 3) * 16, nOff = (wave >> 2) * 32;
    v8f c0 = vzero8(), c1 = vzero8();

    for (int k0 = 0; k0 < CDIM; k0 += 32) {
        __syncthreads();
#pragma unroll
        for (int i = 0; i < 8; ++i) {
            int lin = tid + i * 256;
            int r = lin >> 5, c = lin & 31;
            As[r][c] = f2bf(A[(size_t)(bm + r) * CDIM + k0 + c]);
        }
#pragma unroll
        for (int i = 0; i < 8; ++i) {
            int lin = tid + i * 256;
            int r = lin >> 6, c = lin & 63;
            BsT[c][r] = Wbf[(k0 + r) * CDIM + bn + c];
        }
        __syncthreads();
        v16bf a  = frag_a(&As[mOff][0], 40, lane);
        v16bf b0 = frag_bT(&BsT[nOff][0], 40, lane);
        v16bf b1 = frag_bT(&BsT[nOff + 16][0], 40, lane);
        c0 = wmma_bf16(a, b0, c0);
        c1 = wmma_bf16(a, b1, c1);
    }
    const int m15 = lane & 15, half = lane >> 4;
#pragma unroll
    for (int f = 0; f < 2; ++f)
#pragma unroll
        for (int r = 0; r < 8; ++r) {
            int row = bm + mOff + r + half * 8;
            int col = bn + nOff + f * 16 + m15;
            float v = ((f == 0) ? c0[r] : c1[r]) + bias[col];
            out[(size_t)row * CDIM + col] = v;
        }
}

// ---------------------------------------------------------------------------
// Workspace layout (bytes): ~61.4 MB total
// ---------------------------------------------------------------------------
#define WS_WQKV   ((size_t)0)                       // 768*2304*2
#define WS_WPROJ  (WS_WQKV  + 3538944)              // 768*768*2
#define WS_RHTBL  (WS_WPROJ + 1179648)              // 127*64*2 (padded)
#define WS_RWTBL  (WS_RHTBL + 16384)
#define WS_QBF    (WS_RWTBL + 16384)                // 12*4096*64*2
#define WS_KBF    (WS_QBF   + 6291456)
#define WS_VT     (WS_KBF   + 6291456)              // 12*64*4096*2 (transposed V)
#define WS_RELH   (WS_VT    + 6291456)              // 12*64*4096*4 (transposed)
#define WS_RELW   (WS_RELH  + 12582912)             // 12*4096*64*4
#define WS_OBUF   (WS_RELW  + 12582912)             // 4096*768*4

extern "C" void kernel_launch(void* const* d_in, const int* in_sizes, int n_in,
                              void* d_out, int out_size, void* d_ws, size_t ws_size,
                              hipStream_t stream) {
    (void)in_sizes; (void)n_in; (void)out_size; (void)ws_size;
    const float* X      = (const float*)d_in[0];  // (1,64,64,768)
    const float* qkv_w  = (const float*)d_in[1];  // (768,2304)
    const float* qkv_b  = (const float*)d_in[2];  // (2304,)
    const float* proj_w = (const float*)d_in[3];  // (768,768)
    const float* proj_b = (const float*)d_in[4];  // (768,)
    const float* rph    = (const float*)d_in[5];  // (127,64)
    const float* rpw    = (const float*)d_in[6];  // (127,64)
    float* out = (float*)d_out;

    char* ws = (char*)d_ws;
    unsigned short* Wqkv = (unsigned short*)(ws + WS_WQKV);
    unsigned short* Wprj = (unsigned short*)(ws + WS_WPROJ);
    unsigned short* RhT  = (unsigned short*)(ws + WS_RHTBL);
    unsigned short* RwT  = (unsigned short*)(ws + WS_RWTBL);
    unsigned short* Qbf  = (unsigned short*)(ws + WS_QBF);
    unsigned short* Kbf  = (unsigned short*)(ws + WS_KBF);
    unsigned short* Vt   = (unsigned short*)(ws + WS_VT);
    float* relhT = (float*)(ws + WS_RELH);
    float* relw  = (float*)(ws + WS_RELW);
    float* Obuf  = (float*)(ws + WS_OBUF);

    cvt_bf16_kernel<<<(768 * 2304 + 255) / 256, 256, 0, stream>>>(qkv_w, Wqkv, 768 * 2304);
    cvt_bf16_kernel<<<(768 * 768 + 255) / 256, 256, 0, stream>>>(proj_w, Wprj, 768 * 768);
    cvt_bf16_kernel<<<(127 * 64 + 255) / 256, 256, 0, stream>>>(rph, RhT, 127 * 64);
    cvt_bf16_kernel<<<(127 * 64 + 255) / 256, 256, 0, stream>>>(rpw, RwT, 127 * 64);

    gemm_qkv_kernel<<<dim3(2304 / 64, 4096 / 64), 256, 0, stream>>>(X, Wqkv, qkv_b,
                                                                    Qbf, Kbf, Vt);

    relpos_kernel<<<dim3(64, NHEADS), 256, 0, stream>>>(Qbf, RhT, relhT, 0);
    relpos_kernel<<<dim3(64, NHEADS), 256, 0, stream>>>(Qbf, RwT, relw, 1);

    attn_kernel<<<dim3(4096 / 128, NHEADS), 256, 0, stream>>>(Qbf, Kbf, Vt,
                                                              relhT, relw, Obuf);

    gemm_proj_kernel<<<dim3(768 / 64, 4096 / 64), 256, 0, stream>>>(Obuf, Wprj,
                                                                    proj_b, out);
}
